// NeuralLogicRecAEV2_47510928228669
// MI455X (gfx1250) — matrix operand order repaired and checked by reading, not compiled, not run on hardware
//
#include <hip/hip_runtime.h>
#include <hip/hip_bf16.h>
#include <math.h>

// ---------------------------------------------------------------------------
// NeuralLogicRecAEV2 forward, CDNA5 (gfx1250) WMMA bf16 implementation.
//
// ~262 GFLOP of GEMM vs ~0.75 GB mandatory HBM traffic (32 us @ 23.3 TB/s)
// -> compute bound -> v_wmma_f32_16x16x32_bf16 (bf16 in, f32 accumulate).
// Weights are converted+transposed to bf16 [N,K] once per call so every GEMM
// stages both operands with contiguous 32B loads via the CDNA5 async-to-LDS
// path (ASYNCcnt), double-buffered against the WMMA compute. The encoder's
// first GEMM is split-K (gridDim.z=5) so it fills the machine like the rest.
// ---------------------------------------------------------------------------

typedef __bf16 bf16_t;
typedef __attribute__((ext_vector_type(16))) __bf16 v16bf;
typedef __attribute__((ext_vector_type(8)))  __bf16 v8bf;
typedef __attribute__((ext_vector_type(8)))  float  v8f;

#define RB 2048   // batch
#define RI 20000  // items
#define RD 512    // embedding dim
#define RH 1024   // hidden (2*D)
#define KSPLIT 5  // split-K factor for the first encoder GEMM (20000/5 = 4000)

#define BM 128
#define BN 128
#define BK 32
#define LDSS 40   // LDS row stride in halfs (80 bytes: 16B aligned, bank-spread)

enum {
  EPI_RELU_BF16 = 0,   // outH = bf16(relu(x + bias))
  EPI_SIGMOID_F32 = 1, // outF = sigmoid(x + bias)
  EPI_NEG_F32 = 2,     // outF = -x
  EPI_RELU_DUAL = 3,   // outF = relu(x + bias); outH = bf16(same)
  EPI_PART_F32 = 4     // outF[z*M*N + ix] = x   (split-K partial, no bias)
};

// CDNA5 async copy: LDS[lds_byte_off] = MEM[gptr], 16 bytes, ASYNCcnt-tracked.
__device__ __forceinline__ void async_ld128(unsigned int lds_byte_off,
                                            const bf16_t* gptr) {
  asm volatile("global_load_async_to_lds_b128 %0, %1, off"
               :
               : "v"(lds_byte_off), "v"(gptr)
               : "memory");
}
__device__ __forceinline__ void wait_async0() {
  asm volatile("s_wait_asynccnt 0x0" ::: "memory");
}

// ---------------------------------------------------------------------------
// Tiled WMMA GEMM: C[M,N] = epi(A[M,K] * Bt^T + bias)
//   A : bf16 [M,K] row-major.    Bt: bf16 [N,K] row-major (pre-transposed).
//   Ksub: K-range handled per z-slice (blockIdx.z * Ksub .. +Ksub), Ksub = K
//         and gridDim.z = 1 for everything except the split-K launch.
// Block = 256 threads = 8 waves (2 x 4). Wave tile = 64 x 32 (4x2 WMMA accs).
// Requires M % 128 == 0, Ksub % 32 == 0 (true for all calls). The N edge is
// handled by clamping the staged B row (junk lands only in columns >= N,
// which the guarded C store drops).
// ---------------------------------------------------------------------------
template <int EPI>
__global__ __launch_bounds__(256) void gemm_bt_wmma(
    const bf16_t* __restrict__ A, const bf16_t* __restrict__ Bt,
    const float* __restrict__ bias, float* __restrict__ outF,
    bf16_t* __restrict__ outH, int M, int N, int K, int Ksub) {
  __shared__ __align__(16) bf16_t As[2][BM * LDSS];
  __shared__ __align__(16) bf16_t Bs[2][BN * LDSS];  // stored as [n][k]

  const int tid  = threadIdx.x;
  const int lane = tid & 31;
  const int wave = tid >> 5;
  const int wm   = wave & 1;   // 0..1  -> 64 rows each
  const int wn   = wave >> 1;  // 0..3  -> 32 cols each
  const int kh   = lane >> 4;  // half-wave id (ISA lane-group)
  const int l16  = lane & 15;
  const int bm0  = blockIdx.x * BM;
  const int bn0  = blockIdx.y * BN;
  const size_t koff = (size_t)blockIdx.z * Ksub;

  // staging: each thread owns one 32-byte segment of the A and B tiles
  const int srow = tid >> 1;
  const int sc   = (tid & 1) * 16;
  const bf16_t* aSrc = A + (size_t)(bm0 + srow) * K + koff + sc;
  const int nclamp   = (bn0 + srow < N) ? (bn0 + srow) : (N - 1);
  const bf16_t* bSrc = Bt + (size_t)nclamp * K + koff + sc;
  unsigned int aOff[2], bOff[2];
#pragma unroll
  for (int b = 0; b < 2; ++b) {
    aOff[b] = (unsigned int)(size_t)&As[b][srow * LDSS + sc];
    bOff[b] = (unsigned int)(size_t)&Bs[b][srow * LDSS + sc];
  }

  v8f acc[4][2];
#pragma unroll
  for (int i = 0; i < 4; ++i)
#pragma unroll
    for (int j = 0; j < 2; ++j)
#pragma unroll
      for (int r = 0; r < 8; ++r) acc[i][j][r] = 0.0f;

  const int ktiles = Ksub / BK;
  // prologue: stage tile 0 into buffer 0
  async_ld128(aOff[0], aSrc);
  async_ld128(aOff[0] + 16, aSrc + 8);
  async_ld128(bOff[0], bSrc);
  async_ld128(bOff[0] + 16, bSrc + 8);
  wait_async0();
  __syncthreads();

  // one pipeline stage with a compile-time buffer index
  auto step = [&](int kt, int cur) {
    const int nxt = cur ^ 1;
    if (kt + 1 < ktiles) {  // stage next tile (overlaps with WMMA below)
      const size_t ko = (size_t)(kt + 1) * BK;
      async_ld128(aOff[nxt], aSrc + ko);
      async_ld128(aOff[nxt] + 16, aSrc + ko + 8);
      async_ld128(bOff[nxt], bSrc + ko);
      async_ld128(bOff[nxt] + 16, bSrc + ko + 8);
    }
    // ---- fragments per ISA 7.12.2 layouts ----
    v16bf afrag[4];
#pragma unroll
    for (int tm = 0; tm < 4; ++tm) {
      const int r = wm * 64 + tm * 16 + l16;           // A row = lane%16
      const bf16_t* p = &As[cur][r * LDSS + kh * 8];
      v8bf lo = *(const v8bf*)(p);                     // K = kh*8 + 0..7
      v8bf hi = *(const v8bf*)(p + 16);                // K = kh*8 + 16..23
      afrag[tm] = __builtin_shufflevector(lo, hi, 0, 1, 2, 3, 4, 5, 6, 7, 8, 9,
                                          10, 11, 12, 13, 14, 15);
    }
    v16bf bfrag[2];
#pragma unroll
    for (int tn = 0; tn < 2; ++tn) {
      const int ncol = wn * 32 + tn * 16 + l16;        // B col = lane%16
      const bf16_t* p = &Bs[cur][ncol * LDSS + kh * 8];  // K = kh*8 + 0..7, kh*8 + 16..23
      v8bf lo = *(const v8bf*)(p);
      v8bf hi = *(const v8bf*)(p + 16);
      bfrag[tn] = __builtin_shufflevector(lo, hi, 0, 1, 2, 3, 4, 5, 6, 7, 8, 9,
                                          10, 11, 12, 13, 14, 15);
    }
#pragma unroll
    for (int tm = 0; tm < 4; ++tm)
#pragma unroll
      for (int tn = 0; tn < 2; ++tn)
        acc[tm][tn] = __builtin_amdgcn_wmma_f32_16x16x32_bf16(
            false, afrag[tm], false, bfrag[tn], (short)0, acc[tm][tn], false,
            false);
    // my async writes for the next tile must land, and all waves must finish
    // reading the current buffer, before buffers flip
    wait_async0();
    __syncthreads();
  };

  for (int kt = 0; kt < ktiles; kt += 2) {
    step(kt, 0);
    if (kt + 1 < ktiles) step(kt + 1, 1);
  }

  // ---- epilogue: C VGPR r -> row (r + 8*kh), col lane%16 ----
#pragma unroll
  for (int tm = 0; tm < 4; ++tm)
#pragma unroll
    for (int tn = 0; tn < 2; ++tn) {
      const int gr0 = bm0 + wm * 64 + tm * 16 + kh * 8;
      const int gc  = bn0 + wn * 32 + tn * 16 + l16;
      if (gc >= N) continue;
      const float bv =
          (EPI == EPI_NEG_F32 || EPI == EPI_PART_F32) ? 0.0f : bias[gc];
#pragma unroll
      for (int r = 0; r < 8; ++r) {
        const float v   = acc[tm][tn][r] + bv;
        const size_t ix = (size_t)(gr0 + r) * N + gc;
        if (EPI == EPI_RELU_BF16) {
          outH[ix] = (bf16_t)fmaxf(v, 0.0f);
        } else if (EPI == EPI_SIGMOID_F32) {
          outF[ix] = 1.0f / (1.0f + expf(-v));
        } else if (EPI == EPI_NEG_F32) {
          outF[ix] = -v;
        } else if (EPI == EPI_RELU_DUAL) {
          const float rl = fmaxf(v, 0.0f);
          outF[ix] = rl;
          outH[ix] = (bf16_t)rl;
        } else {  // EPI_PART_F32: split-K partial
          outF[(size_t)blockIdx.z * M * N + ix] = v;
        }
      }
    }
}

// ---- split-K reduction: out = bf16(relu(sum_s parts[s] + bias)) ----
__global__ void reduce_relu_bf16_kernel(const float* __restrict__ parts,
                                        const float* __restrict__ bias,
                                        bf16_t* __restrict__ out, int MN,
                                        int N, int S) {
  const int i = blockIdx.x * blockDim.x + threadIdx.x;
  if (i >= MN) return;
  float s = 0.0f;
  for (int p = 0; p < S; ++p) s += parts[(size_t)p * MN + i];
  out[i] = (bf16_t)fmaxf(s + bias[i % N], 0.0f);
}

// ---- fp32 [K,N] -> bf16 [N,K] tiled transpose-convert (dims % 32 == 0) ----
__global__ __launch_bounds__(256) void transpose_cvt_kernel(
    const float* __restrict__ in, bf16_t* __restrict__ out, int K, int N) {
  __shared__ bf16_t t[32][33];
  const int k0 = blockIdx.x * 32;
  const int n0 = blockIdx.y * 32;
#pragma unroll
  for (int j = 0; j < 4; ++j) {
    const int k = threadIdx.y + j * 8;
    t[k][threadIdx.x] = (bf16_t)in[(size_t)(k0 + k) * N + n0 + threadIdx.x];
  }
  __syncthreads();
#pragma unroll
  for (int j = 0; j < 4; ++j) {
    const int n = threadIdx.y + j * 8;
    out[(size_t)(n0 + n) * K + k0 + threadIdx.x] = t[threadIdx.x][n];
  }
}

// ---- noisy_likes = where(mask, 0, likes) converted to bf16 ----
__global__ void prep_noisy_kernel(const float* __restrict__ likes,
                                  const unsigned char* __restrict__ mask,
                                  bf16_t* __restrict__ out, size_t n) {
  for (size_t i = (size_t)blockIdx.x * blockDim.x + threadIdx.x; i < n;
       i += (size_t)gridDim.x * blockDim.x)
    out[i] = mask[i] ? (bf16_t)0.0f : (bf16_t)likes[i];
}

// ---- popular = mean(likes, axis=0), one column per thread (coalesced) ----
__global__ void popular_kernel(const float* __restrict__ likes,
                               float* __restrict__ out, int Bn, int In) {
  const int i = blockIdx.x * blockDim.x + threadIdx.x;
  if (i >= In) return;
  float s = 0.0f;
  for (int b = 0; b < Bn; ++b) s += likes[(size_t)b * In + i];
  out[i] = s * (1.0f / (float)Bn);
}

// ---- row L2-normalize (tf.nn.l2_normalize eps) -> bf16 rows ----
__global__ __launch_bounds__(256) void rownorm_kernel(
    const float* __restrict__ embed, bf16_t* __restrict__ normed, int Dn) {
  const int row = blockIdx.x;
  const float* src = embed + (size_t)row * Dn;
  float s = 0.0f;
  for (int i = threadIdx.x; i < Dn; i += blockDim.x) {
    const float v = src[i];
    s += v * v;
  }
#pragma unroll
  for (int off = 16; off; off >>= 1) s += __shfl_down(s, off);
  __shared__ float red[8];
  __shared__ float scale_s;
  if ((threadIdx.x & 31) == 0) red[threadIdx.x >> 5] = s;
  __syncthreads();
  if (threadIdx.x == 0) {
    float t = 0.0f;
#pragma unroll
    for (int w = 0; w < 8; ++w) t += red[w];
    scale_s = rsqrtf(fmaxf(t, 1e-12f));
  }
  __syncthreads();
  const float sc = scale_s;
  for (int i = threadIdx.x; i < Dn; i += blockDim.x)
    normed[(size_t)row * Dn + i] = (bf16_t)(src[i] * sc);
}

static inline dim3 gemm_grid(int M, int N, int z = 1) {
  return dim3(M / BM, (N + BN - 1) / BN, z);
}

extern "C" void kernel_launch(void* const* d_in, const int* in_sizes, int n_in,
                              void* d_out, int out_size, void* d_ws,
                              size_t ws_size, hipStream_t stream) {
  (void)in_sizes; (void)n_in; (void)out_size; (void)ws_size;
  // inputs (setup_inputs order)
  const float*         likes  = (const float*)d_in[1];
  const unsigned char* nmask  = (const unsigned char*)d_in[2];
  const float* enc_w1 = (const float*)d_in[3];
  const float* enc_b1 = (const float*)d_in[4];
  const float* enc_w2 = (const float*)d_in[5];
  const float* enc_b2 = (const float*)d_in[6];
  const float* lk_w1  = (const float*)d_in[7];
  const float* lk_b1  = (const float*)d_in[8];
  const float* lk_w2  = (const float*)d_in[9];
  const float* lk_b2  = (const float*)d_in[10];
  const float* rc_w1  = (const float*)d_in[11];
  const float* rc_b1  = (const float*)d_in[12];
  const float* rc_w2  = (const float*)d_in[13];
  const float* rc_b2  = (const float*)d_in[14];

  float* out = (float*)d_out;
  float* out_est_likes = out;                                  // [B,I]
  float* out_user_sim  = out + (size_t)RB * RI;                // [B,B]
  float* out_est_rec   = out_user_sim + (size_t)RB * RB;       // [B,I]
  float* out_popular   = out_est_rec + (size_t)RB * RI;        // [I]

  // workspace bump allocator (256B aligned)
  char* ws = (char*)d_ws;
  size_t off = 0;
  auto alloc = [&](size_t bytes) {
    off = (off + 255) & ~(size_t)255;
    void* p = ws + off;
    off += bytes;
    return p;
  };
  bf16_t* noisy_b = (bf16_t*)alloc((size_t)RB * RI * 2);
  bf16_t* w1t  = (bf16_t*)alloc((size_t)RI * RH * 2);  // [RH, RI]
  bf16_t* w2t  = (bf16_t*)alloc((size_t)RH * RD * 2);  // [RD, RH]
  bf16_t* lw1t = (bf16_t*)alloc((size_t)RD * RH * 2);  // [RH, RD]
  bf16_t* rw1t = (bf16_t*)alloc((size_t)RD * RH * 2);  // [RH, RD]
  bf16_t* lw2t = (bf16_t*)alloc((size_t)RH * RI * 2);  // [RI, RH]
  bf16_t* rw2t = (bf16_t*)alloc((size_t)RH * RI * 2);  // [RI, RH]
  float*  h_part = (float*)alloc((size_t)KSPLIT * RB * RH * 4);
  bf16_t* h_b  = (bf16_t*)alloc((size_t)RB * RH * 2);
  float*  emb_f = (float*)alloc((size_t)RB * RD * 4);
  bf16_t* emb_b = (bf16_t*)alloc((size_t)RB * RD * 2);
  bf16_t* nrm_b = (bf16_t*)alloc((size_t)RB * RD * 2);
  bf16_t* hl_b  = (bf16_t*)alloc((size_t)RB * RH * 2);
  bf16_t* hr_b  = (bf16_t*)alloc((size_t)RB * RH * 2);

  // ---- prep: noisy likes (bf16), weight transpose-conversions, popular ----
  prep_noisy_kernel<<<4096, 256, 0, stream>>>(likes, nmask, noisy_b,
                                              (size_t)RB * RI);
  const dim3 tb(32, 8);
  transpose_cvt_kernel<<<dim3(RI / 32, RH / 32), tb, 0, stream>>>(enc_w1, w1t, RI, RH);
  transpose_cvt_kernel<<<dim3(RH / 32, RD / 32), tb, 0, stream>>>(enc_w2, w2t, RH, RD);
  transpose_cvt_kernel<<<dim3(RD / 32, RH / 32), tb, 0, stream>>>(lk_w1, lw1t, RD, RH);
  transpose_cvt_kernel<<<dim3(RD / 32, RH / 32), tb, 0, stream>>>(rc_w1, rw1t, RD, RH);
  transpose_cvt_kernel<<<dim3(RH / 32, RI / 32), tb, 0, stream>>>(lk_w2, lw2t, RH, RI);
  transpose_cvt_kernel<<<dim3(RH / 32, RI / 32), tb, 0, stream>>>(rc_w2, rw2t, RH, RI);
  popular_kernel<<<(RI + 255) / 256, 256, 0, stream>>>(likes, out_popular, RB, RI);

  // ---- encoder: split-K GEMM1 (5 x 4000) then reduce, then GEMM2 ----
  gemm_bt_wmma<EPI_PART_F32><<<gemm_grid(RB, RH, KSPLIT), 256, 0, stream>>>(
      noisy_b, w1t, nullptr, h_part, nullptr, RB, RH, RI, RI / KSPLIT);
  reduce_relu_bf16_kernel<<<(RB * RH + 255) / 256, 256, 0, stream>>>(
      h_part, enc_b1, h_b, RB * RH, RH, KSPLIT);
  gemm_bt_wmma<EPI_RELU_DUAL><<<gemm_grid(RB, RD), 256, 0, stream>>>(
      h_b, w2t, enc_b2, emb_f, emb_b, RB, RD, RH, RH);

  // ---- cosine-similarity Gram: normalize rows then -(normed @ normed^T) ----
  rownorm_kernel<<<RB, 256, 0, stream>>>(emb_f, nrm_b, RD);
  gemm_bt_wmma<EPI_NEG_F32><<<gemm_grid(RB, RB), 256, 0, stream>>>(
      nrm_b, nrm_b, nullptr, out_user_sim, nullptr, RB, RB, RD, RD);

  // ---- heads ----
  gemm_bt_wmma<EPI_RELU_BF16><<<gemm_grid(RB, RH), 256, 0, stream>>>(
      emb_b, lw1t, lk_b1, nullptr, hl_b, RB, RH, RD, RD);
  gemm_bt_wmma<EPI_RELU_BF16><<<gemm_grid(RB, RH), 256, 0, stream>>>(
      emb_b, rw1t, rc_b1, nullptr, hr_b, RB, RH, RD, RD);
  gemm_bt_wmma<EPI_SIGMOID_F32><<<gemm_grid(RB, RI), 256, 0, stream>>>(
      hl_b, lw2t, lk_b2, out_est_likes, nullptr, RB, RI, RH, RH);
  gemm_bt_wmma<EPI_SIGMOID_F32><<<gemm_grid(RB, RI), 256, 0, stream>>>(
      hr_b, rw2t, rc_b2, out_est_rec, nullptr, RB, RI, RH, RH);
}
